// TransitionDown_68281390072568
// MI455X (gfx1250) — matrix operand order repaired and checked by reading, not compiled, not run on hardware
//
#include <hip/hip_runtime.h>
#include <hip/hip_bf16.h>

// ---- problem constants (match setup_inputs: B=4, N=16384, C=64, O=128) ----
#define B_   4
#define N_   16384
#define C_   64
#define O_   128
#define S_   4096          // npoint = N * 0.25
#define KNB  32            // k
#define K1P  96            // C+3=67 padded to multiple of 32 for f16 WMMA
#define KT1  3             // K-tiles GEMM1 (96/32)
#define KT2  4             // K-tiles GEMM2 (128/32)
#define NT   8             // N-tiles (128/16)

typedef __attribute__((ext_vector_type(16))) _Float16 v16h;
typedef __attribute__((ext_vector_type(8)))  float    v8f;

// Inverse of the 16-bit A-matrix (16x32) WMMA lane layout (contiguous k order):
// element (m, kin) -> lane = m + 16*(kin/16), halfidx = kin%16.
// (B uses kbase = hs?16:0, k = kbase + halfidx.)
__device__ __forceinline__ void a_frag_pos(int m, int kin, int& lane, int& half) {
    lane = m + ((kin >> 4) << 4);
    half = kin & 15;
}

// ----------------------------------------------------------------------------
// 0) Pre-swizzle W1 (padded to K=96) and W2 into WMMA B-fragment-major f16:
//    layout [kt][nt][lane][16 halves]; lane: n = nt*16 + (lane&15),
//    k = kt*32 + (lane>=16 ? 16 : 0) + halfidx.
// ----------------------------------------------------------------------------
__global__ void td_convert_weights(const float* __restrict__ W1,
                                   const float* __restrict__ W2,
                                   _Float16* __restrict__ W1f,
                                   _Float16* __restrict__ W2f) {
    const int tid = blockIdx.x * blockDim.x + threadIdx.x;
    const int stride = gridDim.x * blockDim.x;
    for (int i = tid; i < KT1 * NT * 32 * 16; i += stride) {
        int e = i & 15, lane = (i >> 4) & 31, nt = (i >> 9) & 7, kt = i >> 12;
        int k = kt * 32 + ((lane >> 4) ? 16 : 0) + e;
        int n = nt * 16 + (lane & 15);
        W1f[i] = (k < C_ + 3) ? (_Float16)W1[k * O_ + n] : (_Float16)0.0f;
    }
    for (int i = tid; i < KT2 * NT * 32 * 16; i += stride) {
        int e = i & 15, lane = (i >> 4) & 31, nt = (i >> 9) & 7, kt = i >> 12;
        int k = kt * 32 + ((lane >> 4) ? 16 : 0) + e;
        int n = nt * 16 + (lane & 15);
        W2f[i] = (_Float16)W2[k * O_ + n];
    }
}

// ----------------------------------------------------------------------------
// 1) Farthest point sampling: one workgroup per batch, 1024 threads,
//    16 points/thread in registers; shfl+LDS argmax tree per step.
// ----------------------------------------------------------------------------
#define FPS_T 1024
#define PPT   (N_ / FPS_T)   // 16

__global__ void __launch_bounds__(FPS_T)
td_fps_kernel(const float* __restrict__ xyz, float* __restrict__ new_xyz) {
    const int b = blockIdx.x;
    const int t = threadIdx.x;
    const float* X = xyz + (size_t)b * N_ * 3;

    float px[PPT], py[PPT], pz[PPT], dist[PPT];
#pragma unroll
    for (int j = 0; j < PPT; ++j) {
        int p = t + j * FPS_T;
        px[j] = X[p * 3 + 0];
        py[j] = X[p * 3 + 1];
        pz[j] = X[p * 3 + 2];
        dist[j] = 1e10f;
    }

    __shared__ float redv[32];
    __shared__ int   redi[32];
    __shared__ float sc[3];
    __shared__ int   sFar;
    if (t == 0) sFar = 0;
    __syncthreads();

    for (int i = 0; i < S_; ++i) {
        int far = sFar;
        if (t == 0) {
            float cx = X[far * 3 + 0], cy = X[far * 3 + 1], cz = X[far * 3 + 2];
            sc[0] = cx; sc[1] = cy; sc[2] = cz;
            float* o = new_xyz + ((size_t)b * S_ + i) * 3;
            o[0] = cx; o[1] = cy; o[2] = cz;
        }
        __syncthreads();
        float cx = sc[0], cy = sc[1], cz = sc[2];

        float bv = -1.0f; int bi = 0;
#pragma unroll
        for (int j = 0; j < PPT; ++j) {
            float dx = px[j] - cx, dy = py[j] - cy, dz = pz[j] - cz;
            float d = dx * dx + dy * dy + dz * dz;
            dist[j] = fminf(dist[j], d);
            if (dist[j] > bv) { bv = dist[j]; bi = t + j * FPS_T; }
        }
#pragma unroll
        for (int off = 16; off > 0; off >>= 1) {
            float ov = __shfl_down(bv, off, 32);
            int   oi = __shfl_down(bi, off, 32);
            if (ov > bv || (ov == bv && oi < bi)) { bv = ov; bi = oi; }
        }
        int wid = t >> 5;
        if ((t & 31) == 0) { redv[wid] = bv; redi[wid] = bi; }
        __syncthreads();
        if (wid == 0) {
            bv = redv[t & 31]; bi = redi[t & 31];
#pragma unroll
            for (int off = 16; off > 0; off >>= 1) {
                float ov = __shfl_down(bv, off, 32);
                int   oi = __shfl_down(bi, off, 32);
                if (ov > bv || (ov == bv && oi < bi)) { bv = ov; bi = oi; }
            }
            if (t == 0) sFar = bi;
        }
        __syncthreads();
    }
}

// ----------------------------------------------------------------------------
// 2) kNN over new_xyz: block caches all 4096 sampled points in LDS (48 KB);
//    one thread per query, stable insertion-sorted top-32.
// ----------------------------------------------------------------------------
__global__ void __launch_bounds__(256)
td_knn_kernel(const float* __restrict__ new_xyz, int* __restrict__ knn_idx) {
    const int b = blockIdx.y;
    __shared__ float sx[S_], sy[S_], sz[S_];
    const float* P = new_xyz + (size_t)b * S_ * 3;
    for (int i = threadIdx.x; i < S_; i += blockDim.x) {
        sx[i] = P[i * 3 + 0]; sy[i] = P[i * 3 + 1]; sz[i] = P[i * 3 + 2];
    }
    __syncthreads();

    const int s = blockIdx.x * blockDim.x + threadIdx.x;
    const float qx = sx[s], qy = sy[s], qz = sz[s];

    float dbest[KNB]; int ibest[KNB];
#pragma unroll
    for (int j = 0; j < KNB; ++j) { dbest[j] = 3.4e38f; ibest[j] = 0; }

    for (int j = 0; j < S_; ++j) {
        float dx = sx[j] - qx, dy = sy[j] - qy, dz = sz[j] - qz;
        float d = dx * dx + dy * dy + dz * dz;
        if (d < dbest[KNB - 1]) {
            int pos = KNB - 1;
            while (pos > 0 && dbest[pos - 1] > d) {   // stable ties: earlier index first
                dbest[pos] = dbest[pos - 1];
                ibest[pos] = ibest[pos - 1];
                --pos;
            }
            dbest[pos] = d; ibest[pos] = j;
        }
    }
    int* out = knn_idx + ((size_t)b * S_ + s) * KNB;
#pragma unroll
    for (int j = 0; j < KNB; ++j) out[j] = ibest[j];
}

// ----------------------------------------------------------------------------
// 3) Fused gather + MLP, one workgroup (2 wave32) per sampled point.
//    A staged fragment-major in LDS (contiguous v16h per lane -> ds_load_b128);
//    B fragments read straight from pre-swizzled global weights
//    (contiguous v16h per lane -> global_load_b128, L2-resident).
//    LDS arena 30 KB: {A1f, hb} -> {A2f (disjoint), Db (reuses A1f)}.
// ----------------------------------------------------------------------------
__global__ void __launch_bounds__(64)
td_mlp_kernel(const float* __restrict__ xyz, const float* __restrict__ feat,
              const _Float16* __restrict__ W1f, const _Float16* __restrict__ W2f,
              const float* __restrict__ b1, const float* __restrict__ b2,
              const float* __restrict__ gam, const float* __restrict__ bet,
              const float* __restrict__ new_xyz, const int* __restrict__ knn_idx,
              float* __restrict__ out_feat) {
    const int gb = blockIdx.x;
    const int b = gb / S_, s = gb % S_;
    const int t = threadIdx.x;
    const int lane = t & 31, wv = t >> 5;   // wv = M-tile
    const int ln = lane & 15, hs = lane >> 4;

    __shared__ __align__(32) char smem[30720];
    _Float16* A1f = (_Float16*)(smem);          // [2][3][32][16] halves (6144 B)
    float*    hb  = (float*)(smem + 6144);      // [32][128] f32     (16384 B)
    _Float16* A2f = (_Float16*)(smem + 22528);  // [2][4][32][16] halves (8192 B)
    float*    Db  = (float*)(smem);             // [32][128] f32 (reuses A1f+hb tail)

    __builtin_prefetch(W1f, 0, 0);
    __builtin_prefetch(W2f, 0, 0);

    // --- build A1 fragments (gather features + pos-diff, zero-pad K 67..95) ---
    const int* nb = knn_idx + ((size_t)b * S_ + s) * KNB;
    const float* q = new_xyz + ((size_t)b * S_ + s) * 3;
    const float qx = q[0], qy = q[1], qz = q[2];
    for (int e = t; e < KNB * K1P; e += 64) {
        int r = e / K1P, c = e % K1P;
        int j = nb[r];                       // quirk: indexes ORIGINAL xyz/features
        float val;
        if (c < C_) {
            val = feat[((size_t)b * N_ + j) * C_ + c];
        } else if (c < C_ + 3) {
            int d = c - C_;
            val = xyz[((size_t)b * N_ + j) * 3 + d] - (d == 0 ? qx : (d == 1 ? qy : qz));
        } else {
            val = 0.0f;
        }
        int mt = r >> 4, m = r & 15;
        int kt = c >> 5, kin = c & 31;
        int fl, fh; a_frag_pos(m, kin, fl, fh);
        A1f[(((mt * KT1 + kt) << 5) + fl) * 16 + fh] = (_Float16)val;
    }
    __syncthreads();

    // --- GEMM1: h = A1 @ W1 + b1  (A frags hoisted; B frags from global) ---
    {
        const v16h* Af = (const v16h*)A1f;
        const v16h* Bf = (const v16h*)W1f;
        v16h a0 = Af[((wv * KT1 + 0) << 5) + lane];
        v16h a1 = Af[((wv * KT1 + 1) << 5) + lane];
        v16h a2 = Af[((wv * KT1 + 2) << 5) + lane];
        for (int nt = 0; nt < NT; ++nt) {
            v8f acc = {};
            acc = __builtin_amdgcn_wmma_f32_16x16x32_f16(
                false, a0, false, Bf[((0 * NT + nt) << 5) + lane], (short)0, acc, false, false);
            acc = __builtin_amdgcn_wmma_f32_16x16x32_f16(
                false, a1, false, Bf[((1 * NT + nt) << 5) + lane], (short)0, acc, false, false);
            acc = __builtin_amdgcn_wmma_f32_16x16x32_f16(
                false, a2, false, Bf[((2 * NT + nt) << 5) + lane], (short)0, acc, false, false);
            const int n = nt * 16 + ln;
#pragma unroll
            for (int v = 0; v < 8; ++v) {
                int m = wv * 16 + v + (hs ? 8 : 0);
                hb[m * O_ + n] = acc[v] + b1[n];
            }
        }
    }
    __syncthreads();

    // --- LayerNorm over O + ReLU -> f16 A2 fragments (one row per thread) ---
    if (t < KNB) {
        float mu = 0.0f;
        for (int n = 0; n < O_; ++n) mu += hb[t * O_ + n];
        mu *= (1.0f / O_);
        float var = 0.0f;
        for (int n = 0; n < O_; ++n) { float d = hb[t * O_ + n] - mu; var += d * d; }
        var *= (1.0f / O_);
        float rstd = rsqrtf(var + 1e-5f);
        int mt = t >> 4, m = t & 15;
        for (int n = 0; n < O_; ++n) {
            float hv = (hb[t * O_ + n] - mu) * rstd * gam[n] + bet[n];
            int kt = n >> 5, kin = n & 31;
            int fl, fh; a_frag_pos(m, kin, fl, fh);
            A2f[(((mt * KT2 + kt) << 5) + fl) * 16 + fh] = (_Float16)fmaxf(hv, 0.0f);
        }
    }
    __syncthreads();

    // --- GEMM2: D = A2 @ W2 + b2 ---
    {
        const v16h* Af = (const v16h*)A2f;
        const v16h* Bf = (const v16h*)W2f;
        v16h a0 = Af[((wv * KT2 + 0) << 5) + lane];
        v16h a1 = Af[((wv * KT2 + 1) << 5) + lane];
        v16h a2 = Af[((wv * KT2 + 2) << 5) + lane];
        v16h a3 = Af[((wv * KT2 + 3) << 5) + lane];
        for (int nt = 0; nt < NT; ++nt) {
            v8f acc = {};
            acc = __builtin_amdgcn_wmma_f32_16x16x32_f16(
                false, a0, false, Bf[((0 * NT + nt) << 5) + lane], (short)0, acc, false, false);
            acc = __builtin_amdgcn_wmma_f32_16x16x32_f16(
                false, a1, false, Bf[((1 * NT + nt) << 5) + lane], (short)0, acc, false, false);
            acc = __builtin_amdgcn_wmma_f32_16x16x32_f16(
                false, a2, false, Bf[((2 * NT + nt) << 5) + lane], (short)0, acc, false, false);
            acc = __builtin_amdgcn_wmma_f32_16x16x32_f16(
                false, a3, false, Bf[((3 * NT + nt) << 5) + lane], (short)0, acc, false, false);
            const int n = nt * 16 + ln;
#pragma unroll
            for (int v = 0; v < 8; ++v) {
                int m = wv * 16 + v + (hs ? 8 : 0);
                Db[m * O_ + n] = acc[v] + b2[n];
            }
        }
    }
    __syncthreads();

    // --- max over k=32 neighbor rows -> [O] for this sampled point ---
    for (int n = t; n < O_; n += 64) {
        float mx = Db[n];
#pragma unroll 4
        for (int r = 1; r < KNB; ++r) mx = fmaxf(mx, Db[r * O_ + n]);
        out_feat[((size_t)b * S_ + s) * O_ + n] = mx;
    }
}

// ----------------------------------------------------------------------------
extern "C" void kernel_launch(void* const* d_in, const int* in_sizes, int n_in,
                              void* d_out, int out_size, void* d_ws, size_t ws_size,
                              hipStream_t stream) {
    const float* xyz      = (const float*)d_in[0];
    const float* features = (const float*)d_in[1];
    const float* W1       = (const float*)d_in[2];
    const float* b1       = (const float*)d_in[3];
    const float* gam      = (const float*)d_in[4];
    const float* bet      = (const float*)d_in[5];
    const float* W2       = (const float*)d_in[6];
    const float* b2       = (const float*)d_in[7];

    float* out      = (float*)d_out;
    float* out_xyz  = out;                       // [B,S,3]
    float* out_feat = out + (size_t)B_ * S_ * 3; // [B,S,O]

    char* ws = (char*)d_ws;
    int*      knn_idx = (int*)ws;                                        // B*S*K ints
    _Float16* W1f = (_Float16*)(ws + (size_t)B_ * S_ * KNB * sizeof(int));
    _Float16* W2f = W1f + KT1 * NT * 32 * 16;

    td_convert_weights<<<64, 256, 0, stream>>>(W1, W2, W1f, W2f);
    td_fps_kernel<<<B_, FPS_T, 0, stream>>>(xyz, out_xyz);
    td_knn_kernel<<<dim3(S_ / 256, B_), 256, 0, stream>>>(out_xyz, knn_idx);
    td_mlp_kernel<<<B_ * S_, 64, 0, stream>>>(xyz, features, W1f, W2f,
                                              b1, b2, gam, bet,
                                              out_xyz, knn_idx, out_feat);
}
